// CrissCrossAttention_32882269618795
// MI455X (gfx1250) — matrix operand order repaired and checked by reading, not compiled, not run on hardware
//
#include <hip/hip_runtime.h>
#include <hip/hip_bf16.h>

// ---------------------------------------------------------------------------
// CDNA5 (gfx1250, wave32) criss-cross attention.
//   B=4, H=8, C=64, N=128, D=512, hd=64, L=C*N=8192, M=B*L=32768
// LDS tiles are staged directly in WMMA fragment order (fragment read =
// 2x ds_load_b128). GEMMs are double-buffered: stage K-step k+1 while the
// WMMAs consume K-step k (one workgroup barrier per step).
// ---------------------------------------------------------------------------

typedef __attribute__((ext_vector_type(16))) _Float16 v16h;
typedef __attribute__((ext_vector_type(8)))  _Float16 v8h;
typedef __attribute__((ext_vector_type(8)))  float    v8f;

#define WMMA_F16(a, b, c) \
  __builtin_amdgcn_wmma_f32_16x16x32_f16(false, (a), false, (b), (short)0, (c), false, false)

// Fragment layout (per CDNA5 ISA §7.12.2, wave32):
//  A 16x32 f16 : lane = (m&15) + 16*((k>>3)&1) ; e = ((k>>4)&1)*8 + (k&7)
//  B 32x16 f16 : lane = (n&15) + 16*((k>>3)&1) ; e = ((k>>4)&1)*8 + (k&7)
//  C/D 16x16 f32: VGPR r -> row (r + 8*(lane>>4)), col (lane&15)

__device__ __forceinline__ v16h ld_frag(const _Float16* p) {  // 16B-aligned ok
  const v8h lo = *(const v8h*)p;
  const v8h hi = *(const v8h*)(p + 8);
  return __builtin_shufflevector(lo, hi, 0, 1, 2, 3, 4, 5, 6, 7, 8, 9, 10, 11,
                                 12, 13, 14, 15);
}

__device__ __forceinline__ v8h pack8(float4 a, float4 b) {
  v8h p;
  p[0] = (_Float16)a.x; p[1] = (_Float16)a.y; p[2] = (_Float16)a.z; p[3] = (_Float16)a.w;
  p[4] = (_Float16)b.x; p[5] = (_Float16)b.y; p[6] = (_Float16)b.z; p[7] = (_Float16)b.w;
  return p;
}

// ---------------------------------------------------------------------------
// Kernel 1: qkv[M,1536](f16) = x[M,512](f32) @ w_qkv[512,1536](f32)
// Block tile 128x64, wave tile 32x32 (8 waves: 4m x 2n), K-step 32,
// double-buffered LDS.
// ---------------------------------------------------------------------------
__global__ __launch_bounds__(256) void qkv_gemm_kernel(
    const float* __restrict__ X, const float* __restrict__ W,
    _Float16* __restrict__ QKV) {
  constexpr int BM = 128, BN = 64, BK = 32, K = 512, NN = 1536;
  constexpr int ABUF = (BM / 16) * 32 * 16;  // halves per A buffer
  constexpr int BBUF = (BN / 16) * 32 * 16;  // halves per B buffer
  __shared__ v16h Afrag[2][BM / 16][32];
  __shared__ v16h Bfrag[2][BN / 16][32];

  const int m0 = blockIdx.y * BM, n0 = blockIdx.x * BN;
  const int tid = threadIdx.x, lane = tid & 31, wave = tid >> 5;
  const int wm = wave & 3, wn = wave >> 2;

  // --- per-thread staging geometry (loop-invariant) ---
  const int am = tid >> 2, akb = tid & 3;  // A: 8 consecutive k per thread
  const float* aS0 = &X[(size_t)(m0 + am) * K + akb * 8];
  const float* aS1 = &X[(size_t)(m0 + am + 64) * K + akb * 8];
  _Float16* aD0 = (_Float16*)&Afrag[0][am >> 4][(am & 15) + 16 * (akb & 1)] +
                  ((akb >> 1) & 1) * 8;
  _Float16* aD1 =
      (_Float16*)&Afrag[0][(am + 64) >> 4][(am & 15) + 16 * (akb & 1)] +
      ((akb >> 1) & 1) * 8;
  const int bn = tid & 63, bkb = tid >> 6;
  const float* bS = &W[(size_t)(bkb * 8) * NN + n0 + bn];
  _Float16* bD =
      (_Float16*)&Bfrag[0][bn >> 4][(bn & 15) + 16 * (bkb & 1)] +
      ((bkb >> 1) & 1) * 8;

#define QKV_STAGE(buf, k0)                                                   \
  {                                                                          \
    const float* s0 = aS0 + (k0);                                            \
    __builtin_prefetch(s0 + BK, 0, 1); /* speculative; safe past end */      \
    *(v8h*)(aD0 + (buf)*ABUF) =                                              \
        pack8(*(const float4*)s0, *(const float4*)(s0 + 4));                 \
    const float* s1 = aS1 + (k0);                                            \
    __builtin_prefetch(s1 + BK, 0, 1);                                       \
    *(v8h*)(aD1 + (buf)*ABUF) =                                              \
        pack8(*(const float4*)s1, *(const float4*)(s1 + 4));                 \
    const float* sb = bS + (size_t)(k0)*NN;                                  \
    v8h p;                                                                   \
    for (int t = 0; t < 8; ++t) p[t] = (_Float16)sb[(size_t)t * NN];         \
    *(v8h*)(bD + (buf)*BBUF) = p;                                            \
  }

  v8f acc[2][2] = {{{}, {}}, {{}, {}}};
  QKV_STAGE(0, 0);
  __syncthreads();
  for (int k0 = 0, buf = 0; k0 < K; k0 += BK, buf ^= 1) {
    if (k0 + BK < K) QKV_STAGE(buf ^ 1, k0 + BK);  // overlap with WMMAs
    const v16h a0 = Afrag[buf][wm * 2][lane], a1 = Afrag[buf][wm * 2 + 1][lane];
    const v16h b0 = Bfrag[buf][wn * 2][lane], b1 = Bfrag[buf][wn * 2 + 1][lane];
    acc[0][0] = WMMA_F16(a0, b0, acc[0][0]);
    acc[0][1] = WMMA_F16(a0, b1, acc[0][1]);
    acc[1][0] = WMMA_F16(a1, b0, acc[1][0]);
    acc[1][1] = WMMA_F16(a1, b1, acc[1][1]);
    __syncthreads();
  }
#undef QKV_STAGE
  const int nc = lane & 15, mh = (lane >> 4) << 3;
#pragma unroll
  for (int sm = 0; sm < 2; ++sm)
#pragma unroll
    for (int sn = 0; sn < 2; ++sn)
#pragma unroll
      for (int r = 0; r < 8; ++r) {
        const size_t m = (size_t)m0 + wm * 32 + sm * 16 + r + mh;
        const size_t n = (size_t)n0 + wn * 32 + sn * 16 + nc;
        QKV[m * NN + n] = (_Float16)acc[sm][sn][r];
      }
}

// ---------------------------------------------------------------------------
// Attention core.  MT query rows, NC context rows, hd = 64.
// QKV row l: 1536 halves = [ q(512) | k(512) | v(512) ], head col = h*64.
// ---------------------------------------------------------------------------
template <int MT, int NC, bool ACCUM>
__device__ __forceinline__ void attn_core(
    const _Float16* __restrict__ QKV, float* __restrict__ OUT,
    size_t qRow0, size_t qStride, size_t kRow0, size_t kStride,
    int colQ, size_t out0, size_t outStride) {
  __shared__ v16h Qfrag[MT / 16][2][32];        // A frags for S
  __shared__ v16h Ktfrag[NC / 16][2][32];       // B frags for S (K^T)
  __shared__ _Float16 Vcol[64][NC + 8];         // V column-major: Vcol[d][j]
  __shared__ float Ss[MT][NC];
  __shared__ v16h Pfrag[MT / 16][NC / 32][32];  // A frags for O

  const int tid = threadIdx.x, lane = tid & 31, wave = tid >> 5;
  const int colK = colQ + 512, colV = colQ + 1024;

  // ---- stage Q into A-fragment layout (one b128 load + b128 store each) ----
#pragma unroll
  for (int it = 0; it < MT / 32; ++it) {
    const int i = tid + it * 256;
    const int r = i >> 3, db = i & 7;  // 8 consecutive d
    const v8h q = *(const v8h*)&QKV[(qRow0 + (size_t)r * qStride) * 1536 +
                                    colQ + db * 8];
    *(v8h*)((_Float16*)&Qfrag[r >> 4][db >> 2][(r & 15) + 16 * (db & 1)] +
            ((db >> 1) & 1) * 8) = q;
  }
  // ---- stage K (B-fragment layout) and V (column-major) ----
#pragma unroll
  for (int it = 0; it < NC / 32; ++it) {
    const int i = tid + it * 256;
    const int j = i >> 3, db = i & 7;
    const size_t rowb = (kRow0 + (size_t)j * kStride) * 1536;
    const v8h k8 = *(const v8h*)&QKV[rowb + colK + db * 8];
    *(v8h*)((_Float16*)&Ktfrag[j >> 4][db >> 2]
                              [(j & 15) + 16 * (db & 1)] +
            ((db >> 1) & 1) * 8) = k8;
    const v8h vv = *(const v8h*)&QKV[rowb + colV + db * 8];
#pragma unroll
    for (int t = 0; t < 8; ++t) Vcol[db * 8 + t][j] = vv[t];
  }
  __syncthreads();

  // ---- S = scale * Q K^T : (MT/16)*(NC/16) == 16 tiles, 2 per wave ----
#pragma unroll
  for (int i = 0; i < 2; ++i) {
    const int t = wave * 2 + i;
    const int tm = t / (NC / 16), tn = t % (NC / 16);
    v8f acc = {};
#pragma unroll
    for (int kk = 0; kk < 2; ++kk)
      acc = WMMA_F16(Qfrag[tm][kk][lane], Ktfrag[tn][kk][lane], acc);
    const int n = lane & 15, mh = (lane >> 4) << 3;
#pragma unroll
    for (int r = 0; r < 8; ++r)
      Ss[tm * 16 + r + mh][tn * 16 + n] = acc[r] * 0.125f;  // 1/sqrt(64)
  }
  __syncthreads();

  // ---- row softmax: G=256/MT threads per row, chunk of 16 each ----
  {
    constexpr int G = 256 / MT;  // 4 (spatial) or 8 (temporal)
    const int row = tid / G, g = tid % G;
    const int j0 = g * 16;
    float ev[16];
    float mx = -1e30f;
#pragma unroll
    for (int t = 0; t < 16; ++t) mx = fmaxf(mx, Ss[row][j0 + t]);
#pragma unroll
    for (int m = G >> 1; m; m >>= 1) mx = fmaxf(mx, __shfl_xor(mx, m, G));
    float sum = 0.f;
#pragma unroll
    for (int t = 0; t < 16; ++t) {
      ev[t] = __expf(Ss[row][j0 + t] - mx);
      sum += ev[t];
    }
#pragma unroll
    for (int m = G >> 1; m; m >>= 1) sum += __shfl_xor(sum, m, G);
    const float inv = 1.f / sum;
#pragma unroll
    for (int t = 0; t < 16; ++t) {
      const int j = j0 + t, k = j & 31;
      _Float16* dst = (_Float16*)&Pfrag[row >> 4][j >> 5]
                                       [(row & 15) + 16 * ((k >> 4) & 1)];
      dst[k & 15] = (_Float16)(ev[t] * inv);
    }
  }
  __syncthreads();

  // ---- O = P V : (MT/16)*4 tiles ----
  constexpr int TPW = (MT / 16) * 4 / 8;  // tiles per wave: 2 or 1
#pragma unroll
  for (int i = 0; i < TPW; ++i) {
    const int t = wave * TPW + i;
    const int tm = t >> 2, tn = t & 3;
    v8f acc = {};
#pragma unroll
    for (int kk = 0; kk < NC / 32; ++kk) {
      const v16h a = Pfrag[tm][kk][lane];
      // B frag from column-major V: lane col d = tn*16+(lane&15); 16 j's.
      const v16h b =
          ld_frag(&Vcol[tn * 16 + (lane & 15)][kk * 32 + 16 * (lane >> 4)]);
      acc = WMMA_F16(a, b, acc);
    }
    const int n = lane & 15, mh = (lane >> 4) << 3;
#pragma unroll
    for (int r = 0; r < 8; ++r) {
      const int m = tm * 16 + r + mh;
      const size_t o = out0 + (size_t)m * outStride + tn * 16 + n;
      OUT[o] = ACCUM ? (OUT[o] + acc[r]) : acc[r];
    }
  }
}

// Spatial: one block per (b,h,n); attend over C=64 channels.
__global__ __launch_bounds__(256) void attn_spatial_kernel(
    const _Float16* __restrict__ QKV, float* __restrict__ ATT) {
  const int idx = blockIdx.x;  // B*H*N = 4096
  const int n = idx & 127, h = (idx >> 7) & 7, b = idx >> 10;
  const size_t row0 = (size_t)b * 8192 + n;  // l = c*128 + n, c = 0
  const size_t out0 = ((((size_t)b * 8 + h) * 64 + 0) * 128 + n) * 64;
  attn_core<64, 64, false>(QKV, ATT, row0, 128, row0, 128, h * 64, out0,
                           (size_t)128 * 64);
}

// Temporal: one block per (b,h,c, 32-row slab); attend over N=128 patches.
__global__ __launch_bounds__(256) void attn_temporal_kernel(
    const _Float16* __restrict__ QKV, float* __restrict__ ATT) {
  const int idx = blockIdx.x;  // B*H*C*4 = 8192
  const int nb = idx & 3, c = (idx >> 2) & 63, h = (idx >> 8) & 7,
            b = idx >> 11;
  const size_t kRow0 = (size_t)b * 8192 + c * 128;
  const size_t qRow0 = kRow0 + nb * 32;
  const size_t out0 = ((((size_t)b * 8 + h) * 64 + c) * 128 + nb * 32) * 64;
  attn_core<32, 128, true>(QKV, ATT, qRow0, 1, kRow0, 1, h * 64, out0, 64);
}

// ---------------------------------------------------------------------------
// Kernel 4: out[M,512](f32) = att_reordered[M,512] @ w_out[512,512] + b_out
//   att is [B,H,C,N,hd]; logical A row m=(b, l=c*128+n), col kk=(h*64+d).
//   Double-buffered like kernel 1.
// ---------------------------------------------------------------------------
__global__ __launch_bounds__(256) void out_proj_kernel(
    const float* __restrict__ ATT, const float* __restrict__ W,
    const float* __restrict__ bias, float* __restrict__ OUT) {
  constexpr int BM = 128, BN = 64, BK = 32, K = 512, NN = 512;
  constexpr int ABUF = (BM / 16) * 32 * 16;
  constexpr int BBUF = (BN / 16) * 32 * 16;
  __shared__ v16h Afrag[2][BM / 16][32];
  __shared__ v16h Bfrag[2][BN / 16][32];

  const int m0 = blockIdx.y * BM, n0 = blockIdx.x * BN;
  const int tid = threadIdx.x, lane = tid & 31, wave = tid >> 5;
  const int wm = wave & 3, wn = wave >> 2;

  // A staging geometry: thread owns rows am, am+64; 8 consecutive kk.
  const int am = tid >> 2, akb = tid & 3;
  const int gm0 = m0 + am, gm1 = m0 + am + 64;
  // fixed part of the [B,H,C,N,hd] gather (h,d vary with k0)
  const size_t fix0 = ((size_t)(gm0 >> 13) * 8 * 64 + ((gm0 & 8191) >> 7)) *
                          8192 + (size_t)(gm0 & 127) * 64;
  const size_t fix1 = ((size_t)(gm1 >> 13) * 8 * 64 + ((gm1 & 8191) >> 7)) *
                          8192 + (size_t)(gm1 & 127) * 64;
  _Float16* aD0 = (_Float16*)&Afrag[0][am >> 4][(am & 15) + 16 * (akb & 1)] +
                  ((akb >> 1) & 1) * 8;
  _Float16* aD1 =
      (_Float16*)&Afrag[0][(am + 64) >> 4][(am & 15) + 16 * (akb & 1)] +
      ((akb >> 1) & 1) * 8;
  const int bn = tid & 63, bkb = tid >> 6;
  const float* bS = &W[(size_t)(bkb * 8) * NN + n0 + bn];
  _Float16* bD =
      (_Float16*)&Bfrag[0][bn >> 4][(bn & 15) + 16 * (bkb & 1)] +
      ((bkb >> 1) & 1) * 8;

  // ATT index for row fix, col kk: fix + h*64*8192 ... = fix + (kk>>6)*524288 + (kk&63)
#define PROJ_STAGE(buf, k0)                                                  \
  {                                                                          \
    const int kk = (k0) + akb * 8; /* 8-run stays inside one head */         \
    const size_t hoff = (size_t)(kk >> 6) * 524288 + (kk & 63);              \
    const float* s0 = &ATT[fix0 + hoff];                                     \
    const float* s1 = &ATT[fix1 + hoff];                                     \
    *(v8h*)(aD0 + (buf)*ABUF) =                                              \
        pack8(*(const float4*)s0, *(const float4*)(s0 + 4));                 \
    *(v8h*)(aD1 + (buf)*ABUF) =                                              \
        pack8(*(const float4*)s1, *(const float4*)(s1 + 4));                 \
    const float* sb = bS + (size_t)(k0)*NN;                                  \
    v8h p;                                                                   \
    for (int t = 0; t < 8; ++t) p[t] = (_Float16)sb[(size_t)t * NN];         \
    *(v8h*)(bD + (buf)*BBUF) = p;                                            \
  }

  v8f acc[2][2] = {{{}, {}}, {{}, {}}};
  PROJ_STAGE(0, 0);
  __syncthreads();
  for (int k0 = 0, buf = 0; k0 < K; k0 += BK, buf ^= 1) {
    if (k0 + BK < K) PROJ_STAGE(buf ^ 1, k0 + BK);
    const v16h a0 = Afrag[buf][wm * 2][lane], a1 = Afrag[buf][wm * 2 + 1][lane];
    const v16h b0 = Bfrag[buf][wn * 2][lane], b1 = Bfrag[buf][wn * 2 + 1][lane];
    acc[0][0] = WMMA_F16(a0, b0, acc[0][0]);
    acc[0][1] = WMMA_F16(a0, b1, acc[0][1]);
    acc[1][0] = WMMA_F16(a1, b0, acc[1][0]);
    acc[1][1] = WMMA_F16(a1, b1, acc[1][1]);
    __syncthreads();
  }
#undef PROJ_STAGE
  const int nc = lane & 15, mh = (lane >> 4) << 3;
#pragma unroll
  for (int sm = 0; sm < 2; ++sm)
#pragma unroll
    for (int sn = 0; sn < 2; ++sn)
#pragma unroll
      for (int r = 0; r < 8; ++r) {
        const size_t m = (size_t)m0 + wm * 32 + sm * 16 + r + mh;
        const int n = n0 + wn * 32 + sn * 16 + nc;
        OUT[m * NN + n] = acc[sm][sn][r] + bias[n];
      }
}

// ---------------------------------------------------------------------------
extern "C" void kernel_launch(void* const* d_in, const int* in_sizes, int n_in,
                              void* d_out, int out_size, void* d_ws,
                              size_t ws_size, hipStream_t stream) {
  const float* x     = (const float*)d_in[0];  // [4, 8192, 512]
  const float* w_qkv = (const float*)d_in[1];  // [512, 1536]
  const float* w_out = (const float*)d_in[2];  // [512, 512]
  const float* b_out = (const float*)d_in[3];  // [512]
  float* out = (float*)d_out;                  // [4, 8192, 512]

  _Float16* qkv = (_Float16*)d_ws;  // 96 MB f16
  float* att = (float*)((char*)d_ws + (size_t)32768 * 1536 * sizeof(_Float16));

  const dim3 blk(256);
  qkv_gemm_kernel<<<dim3(1536 / 64, 32768 / 128), blk, 0, stream>>>(x, w_qkv,
                                                                    qkv);
  attn_spatial_kernel<<<dim3(4096), blk, 0, stream>>>(qkv, att);
  attn_temporal_kernel<<<dim3(8192), blk, 0, stream>>>(qkv, att);  // += out_s
  out_proj_kernel<<<dim3(512 / 64, 32768 / 128), blk, 0, stream>>>(att, w_out,
                                                                   b_out, out);
}
